// TransformerLayer_5042291605630
// MI455X (gfx1250) — matrix operand order repaired and checked by reading, not compiled, not run on hardware
//
#include <hip/hip_runtime.h>
#include <math.h>

// ---------------------------------------------------------------------------
// Types for CDNA5 WMMA (wave32): v_wmma_f32_16x16x32_bf16
// ---------------------------------------------------------------------------
typedef __attribute__((ext_vector_type(16))) __bf16 v16bf;
typedef __attribute__((ext_vector_type(8)))  __bf16 bf16x8;
typedef __attribute__((ext_vector_type(4)))  __bf16 bf16x4;
typedef __attribute__((ext_vector_type(8)))  float  v8f;

#ifndef __has_builtin
#define __has_builtin(x) 0
#endif

// gfx1250 async global->LDS copies, emitted via inline asm (the clang builtin
// exists but its pointer params use an unspellable language address space).
#if defined(__HIP_DEVICE_COMPILE__)
#define USE_ASYNC_LDS 1
#else
#define USE_ASYNC_LDS 0
#endif

__device__ __forceinline__ void g2l_16B(const __bf16* g, __bf16* l) {
#if USE_ASYNC_LDS
  // Low 32 bits of a generic pointer to LDS == LDS byte offset (addrspacecast
  // AS3->flat is {aperture_hi, offset}).  VDST = LDS offset, VADDR = 64b addr.
  const unsigned lds_off = (unsigned)(unsigned long long)(const void*)l;
  asm volatile("global_load_async_to_lds_b128 %0, %1, off"
               :
               : "v"(lds_off), "v"(g)
               : "memory");
#else
  *(bf16x8*)l = *(const bf16x8*)g;   // sync fallback: global load + ds_store
#endif
}

template <int N>
__device__ __forceinline__ void async_wait() {
#if USE_ASYNC_LDS
#if __has_builtin(__builtin_amdgcn_s_wait_asynccnt)
  __builtin_amdgcn_s_wait_asynccnt(N);
#else
  asm volatile("s_wait_asynccnt %0" ::"i"(N) : "memory");
#endif
#endif
}

__device__ __forceinline__ v16bf combine16(bf16x8 lo, bf16x8 hi) {
  v16bf r;
#pragma unroll
  for (int i = 0; i < 8; ++i) { r[i] = lo[i]; r[i + 8] = hi[i]; }
  return r;
}

// ---------------------------------------------------------------------------
// Elementwise f32 -> bf16 (weights / activations)
// ---------------------------------------------------------------------------
__global__ __launch_bounds__(256) void cvt_f32_bf16(const float* __restrict__ in,
                                                    __bf16* __restrict__ out, int n4) {
  int i = blockIdx.x * blockDim.x + threadIdx.x;
  const int stride = gridDim.x * blockDim.x;
  for (; i < n4; i += stride) {
    const float4 v = ((const float4*)in)[i];
    bf16x4 o;
    o[0] = (__bf16)v.x; o[1] = (__bf16)v.y; o[2] = (__bf16)v.z; o[3] = (__bf16)v.w;
    ((bf16x4*)out)[i] = o;
  }
}

// ---------------------------------------------------------------------------
// LayerNorm over C=1024, one workgroup (256 threads) per token, bf16 output
// ---------------------------------------------------------------------------
__global__ __launch_bounds__(256) void layernorm_bf16(const float* __restrict__ x,
                                                      const float* __restrict__ g,
                                                      const float* __restrict__ bb,
                                                      __bf16* __restrict__ y) {
  __shared__ float red[8];
  const int row = blockIdx.x;
  const int tid = threadIdx.x;
  const float4 v = *(const float4*)(x + (size_t)row * 1024 + tid * 4);

  float s = v.x + v.y + v.z + v.w;
  s += __shfl_xor(s, 1);  s += __shfl_xor(s, 2);  s += __shfl_xor(s, 4);
  s += __shfl_xor(s, 8);  s += __shfl_xor(s, 16);
  if ((tid & 31) == 0) red[tid >> 5] = s;
  __syncthreads();
  float tot = 0.f;
#pragma unroll
  for (int i = 0; i < 8; ++i) tot += red[i];
  const float mean = tot * (1.0f / 1024.0f);

  const float d0 = v.x - mean, d1 = v.y - mean, d2 = v.z - mean, d3 = v.w - mean;
  float q = d0 * d0 + d1 * d1 + d2 * d2 + d3 * d3;
  q += __shfl_xor(q, 1);  q += __shfl_xor(q, 2);  q += __shfl_xor(q, 4);
  q += __shfl_xor(q, 8);  q += __shfl_xor(q, 16);
  __syncthreads();
  if ((tid & 31) == 0) red[tid >> 5] = q;
  __syncthreads();
  float vq = 0.f;
#pragma unroll
  for (int i = 0; i < 8; ++i) vq += red[i];
  const float rstd = rsqrtf(vq * (1.0f / 1024.0f) + 1e-5f);

  const float4 gg = *(const float4*)(g + tid * 4);
  const float4 be = *(const float4*)(bb + tid * 4);
  bf16x4 o;
  o[0] = (__bf16)(d0 * rstd * gg.x + be.x);
  o[1] = (__bf16)(d1 * rstd * gg.y + be.y);
  o[2] = (__bf16)(d2 * rstd * gg.z + be.z);
  o[3] = (__bf16)(d3 * rstd * gg.w + be.w);
  *(bf16x4*)(y + (size_t)row * 1024 + tid * 4) = o;
}

// ---------------------------------------------------------------------------
// Tiled WMMA GEMM: Y[M,N] = epi( X[M,K] * W[N,K]^T (+resid) )
// 128x128x32 block tile, 8 waves (2x4), 4x2 WMMA tiles per wave,
// double-buffered LDS staged via async global->LDS.
// ---------------------------------------------------------------------------
#define BM 128
#define BN 128
#define BK 32
#define SAP 40  // padded LDS row stride (bf16 elems): 80B rows

__global__ __launch_bounds__(256) void gemm_bf16_wmma(
    const __bf16* __restrict__ A, const __bf16* __restrict__ W,
    float* __restrict__ outF, __bf16* __restrict__ outB,
    const float* __restrict__ resid, int M, int N, int K, int doGelu) {
  __shared__ __attribute__((aligned(16))) __bf16 sA[2][BM * SAP];
  __shared__ __attribute__((aligned(16))) __bf16 sB[2][BN * SAP];

  const int tid  = threadIdx.x;
  const int lane = tid & 31;
  const int hh   = lane >> 4;   // half of wave
  const int l15  = lane & 15;
  const int wave = tid >> 5;
  const int wm   = wave >> 2;   // 0..1 -> 64 rows each
  const int wn   = wave & 3;    // 0..3 -> 32 cols each
  const int m0   = blockIdx.y * BM;
  const int n0   = blockIdx.x * BN;

  // cooperative copy coords: each thread moves 32B of A and 32B of B per tile
  const int cr = tid >> 1;
  const int cc = (tid & 1) * 16;
  const __bf16* gA = A + (size_t)(m0 + cr) * K + cc;
  const __bf16* gB = W + (size_t)(n0 + cr) * K + cc;

#define STAGE(buf, kt)                                      \
  do {                                                      \
    const __bf16* ga_ = gA + (size_t)(kt)*BK;               \
    const __bf16* gb_ = gB + (size_t)(kt)*BK;               \
    __bf16* la_ = &sA[buf][cr * SAP + cc];                  \
    __bf16* lb_ = &sB[buf][cr * SAP + cc];                  \
    g2l_16B(ga_, la_);  g2l_16B(ga_ + 8, la_ + 8);          \
    g2l_16B(gb_, lb_);  g2l_16B(gb_ + 8, lb_ + 8);          \
  } while (0)

  const v8f zero8 = {0.f, 0.f, 0.f, 0.f, 0.f, 0.f, 0.f, 0.f};
  v8f acc[4][2];
#pragma unroll
  for (int i = 0; i < 4; ++i)
#pragma unroll
    for (int j = 0; j < 2; ++j) acc[i][j] = zero8;

  const int nT = K / BK;
  STAGE(0, 0);
  for (int kt = 0; kt < nT; ++kt) {
    const int cur = kt & 1;
    if (kt + 1 < nT) { STAGE(cur ^ 1, kt + 1); async_wait<4>(); }
    else             { async_wait<0>(); }
    __syncthreads();

    const __bf16* pa = &sA[cur][(wm * 64) * SAP];
    const __bf16* pb = &sB[cur][(wn * 32) * SAP];

    // A frags: lane holds row l15; K octets [8*hh, +8) and [16+8*hh, +8)
    v16bf af[4];
#pragma unroll
    for (int mt = 0; mt < 4; ++mt) {
      const __bf16* p = pa + (mt * 16 + l15) * SAP + hh * 8;
      af[mt] = combine16(*(const bf16x8*)p, *(const bf16x8*)(p + 16));
    }
    // B frags: lane holds column l15; K octets [8*hh, +8) and [16+8*hh, +8)
    v16bf bfr[2];
#pragma unroll
    for (int nt = 0; nt < 2; ++nt) {
      const __bf16* p = pb + (nt * 16 + l15) * SAP + hh * 8;
      bfr[nt] = combine16(*(const bf16x8*)p, *(const bf16x8*)(p + 16));
    }
#pragma unroll
    for (int mt = 0; mt < 4; ++mt)
#pragma unroll
      for (int nt = 0; nt < 2; ++nt)
        acc[mt][nt] = __builtin_amdgcn_wmma_f32_16x16x32_bf16(
            false, af[mt], false, bfr[nt], (short)0, acc[mt][nt], false, false);
    __syncthreads();
  }
#undef STAGE

  // Epilogue (C layout: reg r -> row 8*hh + r, lane -> col l15)
#pragma unroll
  for (int mt = 0; mt < 4; ++mt)
#pragma unroll
    for (int nt = 0; nt < 2; ++nt) {
      const int n  = n0 + wn * 32 + nt * 16 + l15;
      const int mb = m0 + wm * 64 + mt * 16 + hh * 8;
#pragma unroll
      for (int r = 0; r < 8; ++r) {
        const size_t idx = (size_t)(mb + r) * N + n;
        float v = acc[mt][nt][r];
        if (doGelu) v = 0.5f * v * (1.0f + erff(v * 0.70710678118654752f));
        if (resid)  v += resid[idx];
        if (outF)   outF[idx] = v;
        if (outB)   outB[idx] = (__bf16)v;
      }
    }
}

// ---------------------------------------------------------------------------
// Flash attention: grid (T/128, H, B); 8 waves x 16 q-rows; 32-key tiles.
// qkv rows are [q(1024) | k(1024) | v(1024)], head h occupies cols h*64..h*64+63.
// ---------------------------------------------------------------------------
__global__ __launch_bounds__(256) void attention_wmma(const __bf16* __restrict__ qkv,
                                                      __bf16* __restrict__ outp) {
  __shared__ __attribute__((aligned(16))) __bf16 sK[32 * 72];      // [key][d]
  __shared__ __attribute__((aligned(16))) __bf16 sVT[64 * 40];     // [d][key]
  __shared__ __attribute__((aligned(16))) __bf16 sP[8][16 * 40];   // per-wave P tile

  const int tid = threadIdx.x, lane = tid & 31, wave = tid >> 5;
  const int hh = lane >> 4, l15 = lane & 15;
  const int h = blockIdx.y, b = blockIdx.z;
  const int q0 = blockIdx.x * 128 + wave * 16;
  const size_t base = (size_t)b * 2048;

  // Q fragments (A layout, 16x32 per k-step s over D=64)
  v16bf qf[2];
  {
    const __bf16* qrow = qkv + (base + q0 + l15) * 3072 + h * 64;
#pragma unroll
    for (int s = 0; s < 2; ++s) {
      const __bf16* p = qrow + s * 32 + hh * 8;
      qf[s] = combine16(*(const bf16x8*)p, *(const bf16x8*)(p + 16));
    }
  }

  const v8f zero8 = {0.f, 0.f, 0.f, 0.f, 0.f, 0.f, 0.f, 0.f};
  v8f o[4];
#pragma unroll
  for (int nt = 0; nt < 4; ++nt) o[nt] = zero8;
  float rmax[8], rsum[8];
#pragma unroll
  for (int r = 0; r < 8; ++r) { rmax[r] = -3.4e38f; rsum[r] = 0.f; }

  for (int k0 = 0; k0 < 2048; k0 += 32) {
    {  // cooperative K tile + transposed V tile
      const int i = tid >> 3, d0 = (tid & 7) * 8;
      const __bf16* kr = qkv + (base + k0 + i) * 3072 + 1024 + h * 64 + d0;
      *(bf16x8*)&sK[i * 72 + d0] = *(const bf16x8*)kr;
      const __bf16* vr = qkv + (base + k0 + i) * 3072 + 2048 + h * 64 + d0;
      const bf16x8 vv = *(const bf16x8*)vr;
#pragma unroll
      for (int e = 0; e < 8; ++e) sVT[(d0 + e) * 40 + i] = vv[e];
    }
    __syncthreads();

    // scores: S(16q x 32k) = Q(16x64) * K^T, two 16-key n-tiles, 2 k-steps each
    v8f sc0 = zero8, sc1 = zero8;
#pragma unroll
    for (int s = 0; s < 2; ++s) {
      const __bf16* p0 = &sK[l15 * 72 + s * 32 + hh * 8];
      const v16bf kb0 = combine16(*(const bf16x8*)p0, *(const bf16x8*)(p0 + 16));
      sc0 = __builtin_amdgcn_wmma_f32_16x16x32_bf16(false, qf[s], false, kb0,
                                                    (short)0, sc0, false, false);
      const __bf16* p1 = &sK[(16 + l15) * 72 + s * 32 + hh * 8];
      const v16bf kb1 = combine16(*(const bf16x8*)p1, *(const bf16x8*)(p1 + 16));
      sc1 = __builtin_amdgcn_wmma_f32_16x16x32_bf16(false, qf[s], false, kb1,
                                                    (short)0, sc1, false, false);
    }

    // online softmax (row = 8*hh + r, reduce across the 16 lanes of the half)
    float alph[8];
#pragma unroll
    for (int r = 0; r < 8; ++r) {
      const float a0 = sc0[r] * 0.125f;  // 1/sqrt(64)
      const float a1 = sc1[r] * 0.125f;
      float mx = fmaxf(a0, a1);
      mx = fmaxf(mx, __shfl_xor(mx, 1));
      mx = fmaxf(mx, __shfl_xor(mx, 2));
      mx = fmaxf(mx, __shfl_xor(mx, 4));
      mx = fmaxf(mx, __shfl_xor(mx, 8));
      const float mn = fmaxf(rmax[r], mx);
      const float al = __expf(rmax[r] - mn);
      rmax[r] = mn;
      const float p0 = __expf(a0 - mn);
      const float p1 = __expf(a1 - mn);
      float ps = p0 + p1;
      ps += __shfl_xor(ps, 1);
      ps += __shfl_xor(ps, 2);
      ps += __shfl_xor(ps, 4);
      ps += __shfl_xor(ps, 8);
      rsum[r] = rsum[r] * al + ps;
      alph[r] = al;
      sP[wave][(hh * 8 + r) * 40 + l15]      = (__bf16)p0;
      sP[wave][(hh * 8 + r) * 40 + 16 + l15] = (__bf16)p1;
    }
#pragma unroll
    for (int nt = 0; nt < 4; ++nt)
#pragma unroll
      for (int r = 0; r < 8; ++r) o[nt][r] *= alph[r];

    // O += P(16x32) * V(32x64); P re-read from LDS in A layout (same wave, LDS in-order)
    const __bf16* pp = &sP[wave][l15 * 40 + hh * 8];
    const v16bf pf = combine16(*(const bf16x8*)pp, *(const bf16x8*)(pp + 16));
#pragma unroll
    for (int nt = 0; nt < 4; ++nt) {
      const __bf16* pv = &sVT[(nt * 16 + l15) * 40 + hh * 8];
      const v16bf vb = combine16(*(const bf16x8*)pv, *(const bf16x8*)(pv + 16));
      o[nt] = __builtin_amdgcn_wmma_f32_16x16x32_bf16(false, pf, false, vb,
                                                      (short)0, o[nt], false, false);
    }
    __syncthreads();
  }

  // normalize + store (C layout)
#pragma unroll
  for (int nt = 0; nt < 4; ++nt)
#pragma unroll
    for (int r = 0; r < 8; ++r) {
      const int tok = q0 + hh * 8 + r;
      const float v = o[nt][r] / rsum[r];
      outp[(base + tok) * 1024 + h * 64 + nt * 16 + l15] = (__bf16)v;
    }
}

// ---------------------------------------------------------------------------
// Host-side orchestration
// ---------------------------------------------------------------------------
extern "C" void kernel_launch(void* const* d_in, const int* in_sizes, int n_in,
                              void* d_out, int out_size, void* d_ws, size_t ws_size,
                              hipStream_t stream) {
  const float* x      = (const float*)d_in[0];
  const float* qkv_w  = (const float*)d_in[1];
  const float* out_w  = (const float*)d_in[2];
  const float* up_w   = (const float*)d_in[3];
  const float* down_w = (const float*)d_in[4];
  const float* ln1_g  = (const float*)d_in[5];
  const float* ln1_b  = (const float*)d_in[6];
  const float* ln2_g  = (const float*)d_in[7];
  const float* ln2_b  = (const float*)d_in[8];

  char* ws = (char*)d_ws;                 // ~120 MB used
  __bf16* Wq  = (__bf16*)(ws + 0);        // 3072x1024 bf16
  __bf16* Wo  = (__bf16*)(ws + 6291456);  // 1024x1024
  __bf16* Wu  = (__bf16*)(ws + 8388608);  // 4096x1024
  __bf16* Wd  = (__bf16*)(ws + 16777216); // 1024x4096
  __bf16* Xb  = (__bf16*)(ws + 25165824); // 4096x1024 (ln1 out)
  __bf16* QKV = (__bf16*)(ws + 33554432); // 4096x3072
  __bf16* ATT = (__bf16*)(ws + 58720256); // 4096x1024
  float*  X1  = (float*) (ws + 67108864); // 4096x1024 f32 (post-attn residual)
  __bf16* H2  = (__bf16*)(ws + 83886080); // 4096x1024 (ln2 out)
  __bf16* UP  = (__bf16*)(ws + 92274688); // 4096x4096 (gelu out)

  cvt_f32_bf16<<<512, 256, 0, stream>>>(qkv_w,  Wq, 3072 * 1024 / 4);
  cvt_f32_bf16<<<512, 256, 0, stream>>>(out_w,  Wo, 1024 * 1024 / 4);
  cvt_f32_bf16<<<512, 256, 0, stream>>>(up_w,   Wu, 4096 * 1024 / 4);
  cvt_f32_bf16<<<512, 256, 0, stream>>>(down_w, Wd, 1024 * 4096 / 4);

  layernorm_bf16<<<4096, 256, 0, stream>>>(x, ln1_g, ln1_b, Xb);
  gemm_bf16_wmma<<<dim3(24, 32), 256, 0, stream>>>(Xb, Wq, nullptr, QKV, nullptr,
                                                   4096, 3072, 1024, 0);
  attention_wmma<<<dim3(16, 16, 2), 256, 0, stream>>>(QKV, ATT);
  gemm_bf16_wmma<<<dim3(8, 32), 256, 0, stream>>>(ATT, Wo, X1, nullptr, x,
                                                  4096, 1024, 1024, 0);
  layernorm_bf16<<<4096, 256, 0, stream>>>(X1, ln2_g, ln2_b, H2);
  gemm_bf16_wmma<<<dim3(32, 32), 256, 0, stream>>>(H2, Wu, nullptr, UP, nullptr,
                                                   4096, 4096, 1024, 1);
  gemm_bf16_wmma<<<dim3(8, 32), 256, 0, stream>>>(UP, Wd, (float*)d_out, nullptr, X1,
                                                  4096, 1024, 4096, 0);
}
